// FEABlock_19516331393114
// MI455X (gfx1250) — matrix operand order repaired and checked by reading, not compiled, not run on hardware
//
#include <hip/hip_runtime.h>
#include <hip/hip_bf16.h>
#include <math.h>

typedef __attribute__((ext_vector_type(16))) __bf16 v16bf;
typedef __attribute__((ext_vector_type(8)))  float  v8f;

#define B_ 32
#define L_ 1024
#define D_ 256
#define H_ 4
#define E_ 64
#define TOPK_ 34

// ---- WMMA operand layouts as packed here (contiguous k order), bf16 16x16x32 ----
// A operand (16 rows x 32 k): value (r16, c) -> lane=((c>>4)&1)*16+r16,
//   elem=c&15
// B operand (32 k x 16 cols): value (k, n16) -> lane=(k>>4)*16+n16, elem=k&15
// Fragments stored as [tile][ks][lane][16] -> contiguous 32B v16bf per lane.

// =====================================================================
// Weight swizzle: W (256x256 fp32) -> bf16 B-fragment layout
// Wswz[nt(16)][ks(8)][lane(32)][e(16)]
// =====================================================================
__global__ void swz_weight_kernel(const float* __restrict__ W, __bf16* __restrict__ Wswz) {
  int i = blockIdx.x * 256 + threadIdx.x;   // 65536 elements
  int e    = i & 15;
  int lane = (i >> 4) & 31;
  int ks   = (i >> 9) & 7;
  int nt   = i >> 12;
  int k = ks * 32 + (lane >> 4) * 16 + e;
  int n = nt * 16 + (lane & 15);
  Wswz[i] = (__bf16)W[k * 256 + n];
}

// =====================================================================
// Activation swizzle: A (Mx256 fp32 row-major) -> bf16 A-fragment layout
// Aswz[mt(M/16)][ks(8)][lane(32)][e(16)]
// =====================================================================
__global__ void swz_act_kernel(const float* __restrict__ A, __bf16* __restrict__ Aswz) {
  int i = blockIdx.x * 256 + threadIdx.x;   // B_*L_*D_ elements
  int e    = i & 15;
  int lane = (i >> 4) & 31;
  int ks   = (i >> 9) & 7;
  int mt   = i >> 12;
  int row = mt * 16 + (lane & 15);
  int k   = ks * 32 + (lane >> 4) * 16 + e;
  Aswz[i] = (__bf16)A[(size_t)row * 256 + k];
}

// =====================================================================
// bf16 WMMA GEMM, fully pre-swizzled operands, no LDS, no barriers.
// C[32768,256] = act(A @ W + bias). Block = 128x64, wave = 32x32 (2x2
// tiles -> 8 global_load_b128 feed 4 WMMAs, 2 loads per matrix op).
// qkv_mode: scatter output to (B,H,E,L): dst[(b*256+col)*1024 + l]
// =====================================================================
__global__ __launch_bounds__(256) void gemm_bf16_kernel(
    const __bf16* __restrict__ Aswz, const __bf16* __restrict__ Wswz,
    const float* __restrict__ bias, float* __restrict__ Cout,
    int act, int qkv_mode) {
  int tid  = threadIdx.x;
  int lane = tid & 31, wave = tid >> 5;
  int half = lane >> 4, l16 = lane & 15;
  int wm = wave & 3;            // row-pair 0..3 (2 tiles of 16 each)
  int wn = wave >> 2;           // col-pair 0..1 (2 tiles of 16 each)
  int m0 = blockIdx.y * 128, n0 = blockIdx.x * 64;
  int mt = blockIdx.y * 8 + wm * 2;
  int nt = blockIdx.x * 4 + wn * 2;

  v8f acc00 = {0.f,0.f,0.f,0.f,0.f,0.f,0.f,0.f};
  v8f acc01 = acc00, acc10 = acc00, acc11 = acc00;

  const __bf16* ap0 = Aswz + (((size_t)mt       * 8) * 32 + lane) * 16;
  const __bf16* ap1 = Aswz + (((size_t)(mt + 1) * 8) * 32 + lane) * 16;
  const __bf16* bp0 = Wswz + (((size_t)nt       * 8) * 32 + lane) * 16;
  const __bf16* bp1 = Wswz + (((size_t)(nt + 1) * 8) * 32 + lane) * 16;
#pragma unroll
  for (int ks = 0; ks < 8; ++ks) {
    const v16bf a0 = *(const v16bf*)(ap0 + ks * 512);
    const v16bf a1 = *(const v16bf*)(ap1 + ks * 512);
    const v16bf b0 = *(const v16bf*)(bp0 + ks * 512);
    const v16bf b1 = *(const v16bf*)(bp1 + ks * 512);
    acc00 = __builtin_amdgcn_wmma_f32_16x16x32_bf16(false, a0, false, b0, (short)0, acc00, false, false);
    acc01 = __builtin_amdgcn_wmma_f32_16x16x32_bf16(false, a0, false, b1, (short)0, acc01, false, false);
    acc10 = __builtin_amdgcn_wmma_f32_16x16x32_bf16(false, a1, false, b0, (short)0, acc10, false, false);
    acc11 = __builtin_amdgcn_wmma_f32_16x16x32_bf16(false, a1, false, b1, (short)0, acc11, false, false);
  }

#pragma unroll
  for (int i = 0; i < 2; ++i) {
#pragma unroll
    for (int j = 0; j < 2; ++j) {
      const v8f& acc = i ? (j ? acc11 : acc10) : (j ? acc01 : acc00);
#pragma unroll
      for (int r = 0; r < 8; ++r) {
        int row = m0 + (wm * 2 + i) * 16 + r + 8 * half;
        int col = n0 + (wn * 2 + j) * 16 + l16;
        float v = acc[r] + bias[col];
        if (act == 1) v = 0.5f * v * (1.0f + erff(v * 0.70710678118f));
        if (qkv_mode) {
          int b = row >> 10, l = row & 1023;
          Cout[((size_t)((b << 8) + col)) * 1024 + l] = v;
        } else {
          Cout[(size_t)row * 256 + col] = v;
        }
      }
    }
  }
}

// =====================================================================
// 1024-pt complex radix-2 Stockham FFT in LDS, table-driven twiddles
// =====================================================================
__device__ void fft1024_s(float2* __restrict__ bufA, float2* __restrict__ bufB,
                          const float2* __restrict__ TW, int tid, bool inv) {
  float2* src = bufA;
  float2* dst = bufB;
  for (int s = 0; s < 10; ++s) {
    int Ns = 1 << s;
    __syncthreads();
    for (int t = tid; t < 512; t += 256) {
      float2 v0 = src[t];
      float2 v1 = src[t + 512];
      int k = t & (Ns - 1);
      float2 w = TW[k << (9 - s)];           // exp(-2*pi*i*k/(2*Ns))
      float sn = inv ? -w.y : w.y;
      float wr = v1.x * w.x - v1.y * sn;
      float wi = v1.x * sn + v1.y * w.x;
      int d = ((t >> s) << (s + 1)) + k;
      dst[d].x = v0.x + wr;  dst[d].y = v0.y + wi;
      dst[d + Ns].x = v0.x - wr;  dst[d + Ns].y = v0.y - wi;
    }
    float2* tmp = src; src = dst; dst = tmp;
  }
  __syncthreads();   // 10 stages (even) -> result back in bufA
}

// =====================================================================
// Per (b,h,e) row: band-pass filter q/k/v + FFT cross-correlation.
// Filtered outputs written straight into WMMA fragment layouts:
//  Qswz[bh][qtile(64)][kk(2)][lane(32)][16]      (A-layout)
//  Kswz[bh][ktile(64)][kk(2)][lane(32)][16]      (B-layout, S = Q.K^T)
//  Vswz[bh][kt(16)][kk(2)][ct(4)][lane(32)][16]  (B-layout, O = P.V)
// =====================================================================
__global__ __launch_bounds__(256) void fft_filter_kernel(
    const float* __restrict__ qT, const float* __restrict__ kT,
    const float* __restrict__ vT, float* __restrict__ corr,
    __bf16* __restrict__ Qswz, __bf16* __restrict__ Kswz, __bf16* __restrict__ Vswz) {
  __shared__ float2 WA[1024];
  __shared__ float2 WB[1024];
  __shared__ float2 SQ[1024];
  __shared__ float2 SK[1024];
  __shared__ float2 SV[1024];
  __shared__ float2 TW[512];
  int tid = threadIdx.x;
  int row = blockIdx.x;                 // b*256 + h*64 + e
  size_t base = (size_t)row * 1024;
  int bh = row >> 6, e = row & 63;

  for (int j = tid; j < 512; j += 256) {
    float ang = -6.283185307179586f * (float)j * (1.f / 1024.f);
    TW[j].x = __cosf(ang);
    TW[j].y = __sinf(ang);
  }

  for (int l = tid; l < 1024; l += 256) { WA[l].x = qT[base + l]; WA[l].y = 0.f; }
  fft1024_s(WA, WB, TW, tid, false);
  for (int l = tid; l < 1024; l += 256) SQ[l] = WA[l];

  __syncthreads();
  for (int l = tid; l < 1024; l += 256) { WA[l].x = kT[base + l]; WA[l].y = 0.f; }
  fft1024_s(WA, WB, TW, tid, false);
  for (int l = tid; l < 1024; l += 256) SK[l] = WA[l];

  __syncthreads();
  for (int l = tid; l < 1024; l += 256) { WA[l].x = vT[base + l]; WA[l].y = 0.f; }
  fft1024_s(WA, WB, TW, tid, false);
  for (int l = tid; l < 1024; l += 256) SV[l] = WA[l];

  // corr = irfft(Q * conj(K) * mask)   (mask symmetric: 205 <= f <= 819)
  __syncthreads();
  for (int f = tid; f < 1024; f += 256) {
    bool m = (f >= 205 && f <= 819);
    float qr = SQ[f].x, qi = SQ[f].y, kr = SK[f].x, ki = SK[f].y;
    WA[f].x = m ? (qr * kr + qi * ki) : 0.f;
    WA[f].y = m ? (qi * kr - qr * ki) : 0.f;
  }
  fft1024_s(WA, WB, TW, tid, true);
  for (int l = tid; l < 1024; l += 256) corr[base + l] = WA[l].x * (1.f / 1024.f);

  // qs -> Qswz (A-layout), channel = e, query index = l, scaled by 1/sqrt(E)/N
  __syncthreads();
  for (int f = tid; f < 1024; f += 256) {
    bool m = (f >= 205 && f <= 819);
    WA[f].x = m ? SQ[f].x : 0.f;  WA[f].y = m ? SQ[f].y : 0.f;
  }
  fft1024_s(WA, WB, TW, tid, true);
  {
    int cc = e & 31;
    int kkq = e >> 5;
    int halfq = (cc >> 4) & 1;
    int elq = cc & 15;
    for (int l = tid; l < 1024; l += 256) {
      size_t idx = (((size_t)(bh * 64 + (l >> 4)) * 2 + kkq) << 9) +
                   halfq * 256 + (l & 15) * 16 + elq;
      Qswz[idx] = (__bf16)(WA[l].x * (0.125f / 1024.f));
    }
  }

  // ks -> Kswz (B-layout), column = key l, k-dim = channel e
  __syncthreads();
  for (int f = tid; f < 1024; f += 256) {
    bool m = (f >= 205 && f <= 819);
    WA[f].x = m ? SK[f].x : 0.f;  WA[f].y = m ? SK[f].y : 0.f;
  }
  fft1024_s(WA, WB, TW, tid, true);
  {
    int kkk = e >> 5;
    int halfk = (e >> 4) & 1;
    int elk = e & 15;
    for (int l = tid; l < 1024; l += 256) {
      size_t idx = (((size_t)(bh * 64 + (l >> 4)) * 2 + kkk) << 9) +
                   halfk * 256 + (l & 15) * 16 + elk;
      Kswz[idx] = (__bf16)(WA[l].x * (1.f / 1024.f));
    }
  }

  // vs -> Vswz (B-layout), k-dim = key l, column = channel e
  __syncthreads();
  for (int f = tid; f < 1024; f += 256) {
    bool m = (f >= 205 && f <= 819);
    WA[f].x = m ? SV[f].x : 0.f;  WA[f].y = m ? SV[f].y : 0.f;
  }
  fft1024_s(WA, WB, TW, tid, true);
  {
    int tnc = e >> 4;
    int n16 = e & 15;
    for (int l = tid; l < 1024; l += 256) {
      int kt = l >> 6, kkv = (l >> 5) & 1, halfv = (l >> 4) & 1, ev = l & 15;
      size_t idx = ((((size_t)(bh * 16 + kt) * 2 + kkv) * 4 + tnc) << 9) +
                   halfv * 256 + n16 * 16 + ev;
      Vswz[idx] = (__bf16)(WA[l].x * (1.f / 1024.f));
    }
  }
}

// =====================================================================
// corr reductions / top-k / softmax weights
// =====================================================================
__global__ __launch_bounds__(256) void corr_mean_kernel(
    const float* __restrict__ corr, float* __restrict__ meanv) {
  int idx = blockIdx.x * 256 + threadIdx.x;   // b*1024 + l
  int b = idx >> 10, l = idx & 1023;
  float s = 0.f;
  for (int i = 0; i < 256; ++i) s += corr[((size_t)((b << 8) + i)) * 1024 + l];
  meanv[idx] = s * (1.f / 256.f);
}

__global__ __launch_bounds__(256) void glob_mean_kernel(
    const float* __restrict__ meanv, float* __restrict__ glob) {
  int l = blockIdx.x * 256 + threadIdx.x;
  float s = 0.f;
  for (int b = 0; b < B_; ++b) s += meanv[b * 1024 + l];
  glob[l] = s * (1.f / (float)B_);
}

__global__ __launch_bounds__(1024) void topk_kernel(
    const float* __restrict__ glob, int* __restrict__ idxout) {
  __shared__ float v[1024];
  __shared__ float rv[1024];
  __shared__ int   ri[1024];
  int t = threadIdx.x;
  v[t] = glob[t];
  __syncthreads();
  for (int j = 0; j < TOPK_; ++j) {
    rv[t] = v[t]; ri[t] = t;
    __syncthreads();
    for (int s = 512; s > 0; s >>= 1) {
      if (t < s && rv[t + s] > rv[t]) { rv[t] = rv[t + s]; ri[t] = ri[t + s]; }
      __syncthreads();
    }
    if (t == 0) { idxout[j] = ri[0]; v[ri[0]] = -3.0e38f; }
    __syncthreads();
  }
}

__global__ void weights_kernel(const float* __restrict__ meanv,
                               const int* __restrict__ idx, float* __restrict__ w) {
  int b = threadIdx.x;
  if (b >= B_) return;
  float vals[TOPK_];
  float mx = -3.0e38f;
  for (int j = 0; j < TOPK_; ++j) {
    float x = meanv[b * 1024 + idx[j]];
    vals[j] = x;
    mx = fmaxf(mx, x);
  }
  float s = 0.f;
  for (int j = 0; j < TOPK_; ++j) { float ex = __expf(vals[j] - mx); vals[j] = ex; s += ex; }
  float inv = 1.f / s;
  for (int j = 0; j < TOPK_; ++j) w[b * 64 + j] = vals[j] * inv;
}

// =====================================================================
// Time-delay aggregation -> writes 0.9 * context_time into ctx (B,L,D)
// =====================================================================
__global__ __launch_bounds__(256) void agg_kernel(
    const float* __restrict__ vT, const float* __restrict__ w,
    const int* __restrict__ idx, float* __restrict__ ctx) {
  __shared__ float vrow[1024];
  __shared__ float wl[TOPK_];
  __shared__ int   il[TOPK_];
  int row = blockIdx.x;            // b*256 + he
  int b = row >> 8, he = row & 255;
  int t = threadIdx.x;
  for (int l = t; l < 1024; l += 256) vrow[l] = vT[(size_t)row * 1024 + l];
  if (t < TOPK_) { wl[t] = w[b * 64 + t]; il[t] = idx[t]; }
  __syncthreads();
  for (int l = t; l < 1024; l += 256) {
    float acc = 0.f;
    for (int j = 0; j < TOPK_; ++j) acc += wl[j] * vrow[(l + il[j]) & 1023];
    ctx[((size_t)((b << 10) + l)) * 256 + he] = 0.9f * acc;
  }
}

// =====================================================================
// Band-filtered attention (two-pass softmax, WMMA): ctx += 0.1 * out
// grid (L/64, B*H), 256 threads (8 waves, 2 accum tiles each).
// Q/K/V fragments stream directly from pre-swizzled global buffers.
// =====================================================================
__global__ __launch_bounds__(256) void attn_kernel(
    const __bf16* __restrict__ Qswz, const __bf16* __restrict__ Kswz,
    const __bf16* __restrict__ Vswz, float* __restrict__ ctx) {
  __shared__ __align__(32) __bf16 Pf[4][2][32][16];  // A-layout (tm, kk)
  __shared__ float  Sst[64][65];
  __shared__ float  rowmax[64];
  __shared__ float  rowsum[64];
  int tid  = threadIdx.x, lane = tid & 31, wave = tid >> 5;
  int half = lane >> 4, l16 = lane & 15;
  int qb = blockIdx.x, bh = blockIdx.y;
  int b = bh >> 2, h = bh & 3;
  int tm = wave & 3, tn0 = (wave >> 2) * 2;

  // Q fragments for this wave's row tile, kept in registers
  const v16bf qf0 = *(const v16bf*)(Qswz + ((((size_t)(bh * 64 + qb * 4 + tm)) * 2 + 0) << 9) + lane * 16);
  const v16bf qf1 = *(const v16bf*)(Qswz + ((((size_t)(bh * 64 + qb * 4 + tm)) * 2 + 1) << 9) + lane * 16);
  if (tid < 64) { rowmax[tid] = -3.0e38f; rowsum[tid] = 0.f; }

  // ---- pass A: row maxima ----
  for (int kt = 0; kt < 16; ++kt) {
    __syncthreads();
#pragma unroll
    for (int t = 0; t < 2; ++t) {
      int tn = tn0 + t;
      const __bf16* kp = Kswz + ((((size_t)(bh * 64 + kt * 4 + tn)) * 2) << 9) + lane * 16;
      v8f s = {0.f,0.f,0.f,0.f,0.f,0.f,0.f,0.f};
      const v16bf k0 = *(const v16bf*)(kp);
      const v16bf k1 = *(const v16bf*)(kp + 512);
      s = __builtin_amdgcn_wmma_f32_16x16x32_bf16(false, qf0, false, k0, (short)0, s, false, false);
      s = __builtin_amdgcn_wmma_f32_16x16x32_bf16(false, qf1, false, k1, (short)0, s, false, false);
#pragma unroll
      for (int r = 0; r < 8; ++r) Sst[tm * 16 + r + 8 * half][tn * 16 + l16] = s[r];
    }
    __syncthreads();
    if (tid < 64) {
      float m = rowmax[tid];
      for (int sx = 0; sx < 64; ++sx) m = fmaxf(m, Sst[tid][sx]);
      rowmax[tid] = m;
    }
  }

  // ---- pass B: exp / sum / P @ V ----
  v8f acc0 = {0.f,0.f,0.f,0.f,0.f,0.f,0.f,0.f};
  v8f acc1 = acc0;
  for (int kt = 0; kt < 16; ++kt) {
    __syncthreads();
#pragma unroll
    for (int t = 0; t < 2; ++t) {
      int tn = tn0 + t;
      const __bf16* kp = Kswz + ((((size_t)(bh * 64 + kt * 4 + tn)) * 2) << 9) + lane * 16;
      v8f s = {0.f,0.f,0.f,0.f,0.f,0.f,0.f,0.f};
      const v16bf k0 = *(const v16bf*)(kp);
      const v16bf k1 = *(const v16bf*)(kp + 512);
      s = __builtin_amdgcn_wmma_f32_16x16x32_bf16(false, qf0, false, k0, (short)0, s, false, false);
      s = __builtin_amdgcn_wmma_f32_16x16x32_bf16(false, qf1, false, k1, (short)0, s, false, false);
#pragma unroll
      for (int r = 0; r < 8; ++r) Sst[tm * 16 + r + 8 * half][tn * 16 + l16] = s[r];
    }
    __syncthreads();
    if (tid < 64) {
      float m = rowmax[tid], rs = 0.f;
      int i16 = tid & 15, itm = tid >> 4;
      for (int sx = 0; sx < 64; ++sx) {
        float p = __expf(Sst[tid][sx] - m);
        rs += p;
        int cc = sx & 31;
        Pf[itm][sx >> 5][((cc >> 4) & 1) * 16 + i16][cc & 15] = (__bf16)p;
      }
      rowsum[tid] += rs;
    }
    __syncthreads();
#pragma unroll
    for (int kk = 0; kk < 2; ++kk) {
      const v16bf a   = *(const v16bf*)Pf[tm][kk][lane];
      const __bf16* vp = Vswz + (((((size_t)(bh * 16 + kt)) * 2 + kk) * 4) << 9) + lane * 16;
      const v16bf b0v = *(const v16bf*)(vp + (size_t)tn0 * 512);
      const v16bf b1v = *(const v16bf*)(vp + (size_t)(tn0 + 1) * 512);
      acc0 = __builtin_amdgcn_wmma_f32_16x16x32_bf16(false, a, false, b0v, (short)0, acc0, false, false);
      acc1 = __builtin_amdgcn_wmma_f32_16x16x32_bf16(false, a, false, b1v, (short)0, acc1, false, false);
    }
  }
  __syncthreads();
#pragma unroll
  for (int r = 0; r < 8; ++r) {
    int ri = tm * 16 + r + 8 * half;
    float inv = 1.f / rowsum[ri];
    int l = qb * 64 + ri;
    size_t o = ((size_t)(b * 1024 + l)) * 256 + h * 64;
    ctx[o + tn0 * 16 + l16]        += 0.1f * acc0[r] * inv;
    ctx[o + (tn0 + 1) * 16 + l16]  += 0.1f * acc1[r] * inv;
  }
}

// =====================================================================
// out = LayerNorm(ya + res) * g + b       (one block per row of D=256)
// =====================================================================
__global__ __launch_bounds__(256) void add_ln_kernel(
    const float* __restrict__ ya, const float* __restrict__ res,
    const float* __restrict__ g, const float* __restrict__ bt,
    float* __restrict__ out) {
  __shared__ float red[256];
  int row = blockIdx.x, t = threadIdx.x;
  float v = ya[(size_t)row * 256 + t] + res[(size_t)row * 256 + t];
  red[t] = v; __syncthreads();
  for (int s = 128; s > 0; s >>= 1) { if (t < s) red[t] += red[t + s]; __syncthreads(); }
  float mean = red[0] * (1.f / 256.f);
  __syncthreads();
  float c = v - mean;
  red[t] = c * c; __syncthreads();
  for (int s = 128; s > 0; s >>= 1) { if (t < s) red[t] += red[t + s]; __syncthreads(); }
  float var = red[0] * (1.f / 256.f);
  out[(size_t)row * 256 + t] = c * rsqrtf(var + 1e-8f) * g[t] + bt[t];
}

// =====================================================================
extern "C" void kernel_launch(void* const* d_in, const int* in_sizes, int n_in,
                              void* d_out, int out_size, void* d_ws, size_t ws_size,
                              hipStream_t stream) {
  (void)in_sizes; (void)n_in; (void)out_size; (void)ws_size;
  const float* x    = (const float*)d_in[0];
  const float* Wq   = (const float*)d_in[1];
  const float* bq   = (const float*)d_in[2];
  const float* Wk   = (const float*)d_in[3];
  const float* bk   = (const float*)d_in[4];
  const float* Wv   = (const float*)d_in[5];
  const float* bv   = (const float*)d_in[6];
  const float* Wd   = (const float*)d_in[7];
  const float* bd   = (const float*)d_in[8];
  const float* ln1g = (const float*)d_in[9];
  const float* ln1b = (const float*)d_in[10];
  const float* W1   = (const float*)d_in[11];
  const float* b1   = (const float*)d_in[12];
  const float* W2   = (const float*)d_in[13];
  const float* b2   = (const float*)d_in[14];
  const float* ln2g = (const float*)d_in[15];
  const float* ln2b = (const float*)d_in[16];
  float* out = (float*)d_out;

  char* wp = (char*)d_ws;
  const size_t SZ = (size_t)B_ * L_ * D_;   // 8,388,608
  auto carve = [&](size_t bytes) -> char* {
    char* p = wp;
    wp += (bytes + 255) & ~(size_t)255;
    return p;
  };
  float*  qT     = (float*)carve(SZ * 4);      // (B,H,E,L)
  float*  kT     = (float*)carve(SZ * 4);
  float*  vT     = (float*)carve(SZ * 4);
  float*  corr   = (float*)carve(SZ * 4);      // (B,H,E,L)
  float*  ctx    = (float*)carve(SZ * 4);      // (B,L,D)
  __bf16* Qswz   = (__bf16*)carve(SZ * 2);     // attention fragment layouts
  __bf16* Kswz   = (__bf16*)carve(SZ * 2);
  __bf16* Vswz   = (__bf16*)carve(SZ * 2);
  __bf16* xswz   = (__bf16*)carve(SZ * 2);     // activation A-fragments
  __bf16* actswz = (__bf16*)carve(SZ * 2);
  float*  meanv  = (float*)carve((size_t)B_ * L_ * 4);
  float*  glob   = (float*)carve(L_ * 4);
  int*    idx    = (int*)carve(64 * 4);
  float*  wts    = (float*)carve((size_t)B_ * 64 * 4);
  __bf16* wq_b   = (__bf16*)carve((size_t)D_ * D_ * 2);
  __bf16* wk_b   = (__bf16*)carve((size_t)D_ * D_ * 2);
  __bf16* wv_b   = (__bf16*)carve((size_t)D_ * D_ * 2);
  __bf16* wd_b   = (__bf16*)carve((size_t)D_ * D_ * 2);
  __bf16* w1_b   = (__bf16*)carve((size_t)D_ * D_ * 2);
  __bf16* w2_b   = (__bf16*)carve((size_t)D_ * D_ * 2);
  // reuse dead buffers for the back half
  float* ybuf = qT;     // Wd GEMM output
  float* hbuf = kT;     // LN1 output
  float* gbuf = corr;   // GELU(FFN1) output
  float* fbuf = vT;     // FFN2 output

  const int M = B_ * L_;               // 32768
  dim3 gemm_grid(D_ / 64, M / 128);    // (4, 256): 128x64 blocks
  const int DD = D_ * D_;
  const int NSWZ = (int)(SZ / 256);    // blocks for activation swizzle

  swz_weight_kernel<<<DD / 256, 256, 0, stream>>>(Wq, wq_b);
  swz_weight_kernel<<<DD / 256, 256, 0, stream>>>(Wk, wk_b);
  swz_weight_kernel<<<DD / 256, 256, 0, stream>>>(Wv, wv_b);
  swz_weight_kernel<<<DD / 256, 256, 0, stream>>>(Wd, wd_b);
  swz_weight_kernel<<<DD / 256, 256, 0, stream>>>(W1, w1_b);
  swz_weight_kernel<<<DD / 256, 256, 0, stream>>>(W2, w2_b);

  // QKV projections -> (B,H,E,L); x swizzled once, reused 3x
  swz_act_kernel<<<NSWZ, 256, 0, stream>>>(x, xswz);
  gemm_bf16_kernel<<<gemm_grid, 256, 0, stream>>>(xswz, wq_b, bq, qT, 0, 1);
  gemm_bf16_kernel<<<gemm_grid, 256, 0, stream>>>(xswz, wk_b, bk, kT, 0, 1);
  gemm_bf16_kernel<<<gemm_grid, 256, 0, stream>>>(xswz, wv_b, bv, vT, 0, 1);

  // FFT band-pass + correlation, outputs pre-swizzled for WMMA
  fft_filter_kernel<<<B_ * H_ * E_, 256, 0, stream>>>(qT, kT, vT, corr, Qswz, Kswz, Vswz);

  // time-delay aggregation path
  corr_mean_kernel<<<(B_ * L_) / 256, 256, 0, stream>>>(corr, meanv);
  glob_mean_kernel<<<L_ / 256, 256, 0, stream>>>(meanv, glob);
  topk_kernel<<<1, 1024, 0, stream>>>(glob, idx);
  weights_kernel<<<1, 32, 0, stream>>>(meanv, idx, wts);
  agg_kernel<<<B_ * H_ * E_, 256, 0, stream>>>(vT, wts, idx, ctx);

  // spatial attention, blended into ctx
  attn_kernel<<<dim3(L_ / 64, B_ * H_), 256, 0, stream>>>(Qswz, Kswz, Vswz, ctx);

  // output projection + LN1
  swz_act_kernel<<<NSWZ, 256, 0, stream>>>(ctx, actswz);
  gemm_bf16_kernel<<<gemm_grid, 256, 0, stream>>>(actswz, wd_b, bd, ybuf, 0, 0);
  add_ln_kernel<<<M, 256, 0, stream>>>(ybuf, x, ln1g, ln1b, hbuf);

  // FFN + LN2
  swz_act_kernel<<<NSWZ, 256, 0, stream>>>(hbuf, actswz);
  gemm_bf16_kernel<<<gemm_grid, 256, 0, stream>>>(actswz, w1_b, b1, gbuf, 1, 0);
  swz_act_kernel<<<NSWZ, 256, 0, stream>>>(gbuf, actswz);
  gemm_bf16_kernel<<<gemm_grid, 256, 0, stream>>>(actswz, w2_b, b2, fbuf, 0, 0);
  add_ln_kernel<<<M, 256, 0, stream>>>(fbuf, hbuf, ln2g, ln2b, out);
}
